// SentenceBertLSTM_63462436765919
// MI455X (gfx1250) — matrix operand order repaired and checked by reading, not compiled, not run on hardware
//
#include <hip/hip_runtime.h>
#include <math.h>

#define DEV __device__ __forceinline__

typedef __bf16 bhalf;
typedef bhalf v16bf __attribute__((ext_vector_type(16)));
typedef bhalf v8bf  __attribute__((ext_vector_type(8)));
typedef float v8f   __attribute__((ext_vector_type(8)));
typedef float v4f   __attribute__((ext_vector_type(4)));

// Problem dims (match reference)
constexpr int Gd   = 16;
constexpr int Ad   = 64;
constexpr int Pd   = 32;
constexpr int DIN  = 768;
constexpr int DEMB = 384;
constexpr int HPd  = 256;
constexpr int HAd  = 128;

constexpr int B1   = Gd * Ad;     // 1024  post-LSTM batch
constexpr int ME   = B1 * Pd;     // 32768 rows for embed GEMM
constexpr int NG1  = 4 * HPd;     // 1024  post gates width
constexpr int NG2  = 4 * HAd;     // 512   author gates width

// ---------------------------------------------------------------------------
// WMMA helpers (wave32, v_wmma_f32_16x16x32_bf16)
// ---------------------------------------------------------------------------
DEV v8f wmma_bf16(v16bf a, v16bf b, v8f c) {
  return __builtin_amdgcn_wmma_f32_16x16x32_bf16(
      /*neg_a=*/false, a, /*neg_b=*/false, b,
      /*c_mod=*/(short)0, c, /*reuse_a=*/false, /*reuse_b=*/false);
}

// A fragment (16x32, bf16, row-major with explicit row stride).
// ISA layout: lanes 0-15 hold row M=lane, K = {0..7, 16..23};
//             lanes 16-31 hold row M=lane-16, K = {8..15, 24..31}.
DEV v16bf load_a_bf16(const bhalf* base, int rowStride, int m0, int k0, int lane) {
  int m  = m0 + (lane & 15);
  int kb = k0 + ((lane >> 4) << 3);
  const bhalf* p = base + (long)m * rowStride + kb;
  v8bf lo = *(const v8bf*)(p);        // K = kb .. kb+7
  v8bf hi = *(const v8bf*)(p + 16);   // K = kb+16 .. kb+23
  v16bf a;
#pragma unroll
  for (int e = 0; e < 8; ++e) { a[e] = lo[e]; a[e + 8] = hi[e]; }
  return a;
}

// A fragment from f32 source (converted to bf16 in-flight) — used for x.
DEV v16bf load_a_f32(const float* base, int rowStride, int m0, int k0, int lane) {
  int m  = m0 + (lane & 15);
  int kb = k0 + ((lane >> 4) << 3);
  const float* p = base + (long)m * rowStride + kb;
  v4f f0 = *(const v4f*)(p);
  v4f f1 = *(const v4f*)(p + 4);
  v4f f2 = *(const v4f*)(p + 16);
  v4f f3 = *(const v4f*)(p + 20);
  // gfx1250 prefetch of next K-block of this row into near caches (WGP scope)
  __builtin_prefetch(p + 32, 0, 3);
  v16bf a;
#pragma unroll
  for (int e = 0; e < 4; ++e) {
    a[e]      = (bhalf)f0[e];
    a[e + 4]  = (bhalf)f1[e];
    a[e + 8]  = (bhalf)f2[e];
    a[e + 12] = (bhalf)f3[e];
  }
  return a;
}

// B fragment from pre-packed fragment-major buffer:
// Bp[((kstep*ntiles + ntile)*32 + lane)*16 + e]
DEV v16bf load_b_pack(const bhalf* Bp, int kstep, int ntile, int ntiles, int lane) {
  const bhalf* p = Bp + ((((long)kstep * ntiles + ntile) * 32) + lane) * 16;
  v8bf lo = *(const v8bf*)(p);
  v8bf hi = *(const v8bf*)(p + 8);
  v16bf b;
#pragma unroll
  for (int e = 0; e < 8; ++e) { b[e] = lo[e]; b[e + 8] = hi[e]; }
  return b;
}

DEV float sigmoidf_(float x) { return 1.0f / (1.0f + __expf(-x)); }

// ---------------------------------------------------------------------------
// Weight pack: f32 [K,N] row-major -> bf16 WMMA B-fragment order.
// Lanes 0-15 hold K={0..7,16..23}, lanes 16-31 hold K={8..15,24..31} of each
// 32-K block; column N = ntile*16 + (lane&15).
// ---------------------------------------------------------------------------
__global__ void pack_b_kernel(const float* __restrict__ B, bhalf* __restrict__ Bp,
                              int K, int N) {
  int idx    = blockIdx.x * blockDim.x + threadIdx.x;
  int ksteps = K >> 5;
  int ntiles = N >> 4;
  int total  = ksteps * ntiles * 32;
  if (idx >= total) return;
  int lane  = idx & 31;
  int ntile = (idx >> 5) % ntiles;
  int kstep = (idx >> 5) / ntiles;
  int n  = ntile * 16 + (lane & 15);
  int kb = kstep * 32 + ((lane >> 4) << 3);
  bhalf* dst = Bp + (long)idx * 16;
#pragma unroll
  for (int e = 0; e < 16; ++e) dst[e] = (bhalf)B[(long)(kb + e + ((e >> 3) << 3)) * N + n];
}

// Zero LSTM states (h ping buffers in bf16, c in f32).
__global__ void init_state_kernel(bhalf* hp0, float* cpost, bhalf* ha0, float* ca) {
  int idx = blockIdx.x * blockDim.x + threadIdx.x;
  if (idx < B1 * HPd) { hp0[idx] = (bhalf)0.0f; cpost[idx] = 0.0f; }
  if (idx < Gd * HAd) { ha0[idx] = (bhalf)0.0f; ca[idx]    = 0.0f; }
}

// ---------------------------------------------------------------------------
// Stage 1: e = x @ w_embed + b_embed   ([32768,768] x [768,384] -> bf16)
// 1x4 register tile per wave. All 4 B fragments are preloaded before the
// WMMA quad so the fragment fetches overlap matrix-core execution instead of
// each WMMA draining loadcnt to 0.
// ---------------------------------------------------------------------------
__global__ __launch_bounds__(256) void embed_gemm_kernel(
    const float* __restrict__ x, const bhalf* __restrict__ wp,
    const float* __restrict__ bias, bhalf* __restrict__ e) {
  constexpr int NT  = DEMB / 16;   // 24 n-tiles
  constexpr int NTG = NT / 4;      // 6 groups of 4
  int wave = blockIdx.x * (blockDim.x >> 5) + (threadIdx.x >> 5);
  int lane = threadIdx.x & 31;
  int mt  = wave / NTG;
  int nt0 = (wave % NTG) * 4;
  v8f acc[4] = {};
#pragma unroll 2
  for (int ks = 0; ks < DIN / 32; ++ks) {      // 24 k-steps
    v16bf a = load_a_f32(x, DIN, mt * 16, ks * 32, lane);
    v16bf bf[4];
#pragma unroll
    for (int u = 0; u < 4; ++u) bf[u] = load_b_pack(wp, ks, nt0 + u, NT, lane);
#pragma unroll
    for (int u = 0; u < 4; ++u) acc[u] = wmma_bf16(a, bf[u], acc[u]);
  }
  int col  = lane & 15;
  int mrow = mt * 16 + ((lane >> 4) << 3);
#pragma unroll
  for (int u = 0; u < 4; ++u) {
    int n    = (nt0 + u) * 16 + col;
    float bv = bias[n];
#pragma unroll
    for (int j = 0; j < 8; ++j)
      e[(long)(mrow + j) * DEMB + n] = (bhalf)(acc[u][j] + bv);
  }
}

// ---------------------------------------------------------------------------
// Stage 2: fused LSTM step (post level), one kernel per timestep.
// Block = 4 waves (128 thr). Wave q computes gate q's 16x16 tile of
//   gates = e[:,t,:] @ k_post + h_in @ r_post        (concatenated-K GEMM)
// at rows mt*16.. and h-columns ncol*16.. ; gate tiles exchanged via LDS,
// then the block applies the LSTM cell to its exclusive 16x16 (b,n) patch.
// K-loops are 2-deep preloaded; h is double-buffered across steps.
// ---------------------------------------------------------------------------
__global__ __launch_bounds__(128) void lstm_post_step_kernel(
    const bhalf* __restrict__ e, const bhalf* __restrict__ hin,
    const bhalf* __restrict__ kp, const bhalf* __restrict__ rp,
    const float* __restrict__ bias, float* __restrict__ cst,
    bhalf* __restrict__ hout, int t) {
  __shared__ float ls[4 * 256];
  int lane = threadIdx.x & 31;
  int q    = threadIdx.x >> 5;             // gate index 0..3 (i,f,c,o)
  int mt   = blockIdx.x >> 4;              // 64 row tiles
  int ncol = blockIdx.x & 15;              // 16 h-column groups
  int nt   = q * (HPd / 16) + ncol;        // n-tile in 4H space

  v8f c = {};
  const bhalf* abase = e + (long)t * DEMB; // row b -> e[(b*P + t)*DEMB]
#pragma unroll 2
  for (int ks = 0; ks < DEMB / 32; ks += 2) {  // 12 k-steps over e
    v16bf a0 = load_a_bf16(abase, Pd * DEMB, mt * 16, ks * 32, lane);
    v16bf b0 = load_b_pack(kp, ks, nt, NG1 / 16, lane);
    v16bf a1 = load_a_bf16(abase, Pd * DEMB, mt * 16, ks * 32 + 32, lane);
    v16bf b1 = load_b_pack(kp, ks + 1, nt, NG1 / 16, lane);
    c = wmma_bf16(a0, b0, c);
    c = wmma_bf16(a1, b1, c);
  }
#pragma unroll 2
  for (int ks = 0; ks < HPd / 32; ks += 2) {   // 8 k-steps over h
    v16bf a0 = load_a_bf16(hin, HPd, mt * 16, ks * 32, lane);
    v16bf b0 = load_b_pack(rp, ks, nt, NG1 / 16, lane);
    v16bf a1 = load_a_bf16(hin, HPd, mt * 16, ks * 32 + 32, lane);
    v16bf b1 = load_b_pack(rp, ks + 1, nt, NG1 / 16, lane);
    c = wmma_bf16(a0, b0, c);
    c = wmma_bf16(a1, b1, c);
  }
  {
    int col   = lane & 15;
    int rbase = (lane >> 4) << 3;
#pragma unroll
    for (int j = 0; j < 8; ++j)
      ls[q * 256 + (rbase + j) * 16 + col] = c[j];
  }
  __syncthreads();
  for (int i = threadIdx.x; i < 256; i += 128) {
    int r   = i >> 4;
    int col = i & 15;
    int bi  = mt * 16 + r;
    int n   = ncol * 16 + col;
    float gi = sigmoidf_(ls[0 * 256 + i] + bias[0 * HPd + n]);
    float gf = sigmoidf_(ls[1 * 256 + i] + bias[1 * HPd + n]);
    float gg = tanhf   (ls[2 * 256 + i] + bias[2 * HPd + n]);
    float go = sigmoidf_(ls[3 * 256 + i] + bias[3 * HPd + n]);
    long idx = (long)bi * HPd + n;
    float cv = gf * cst[idx] + gi * gg;
    cst[idx]  = cv;
    hout[idx] = (bhalf)(go * tanhf(cv));
  }
}

// ---------------------------------------------------------------------------
// Stage 3: fused LSTM step (author level), one kernel per author step.
// M = 16 (= G) rows, gates width 512; 8 blocks x 128 threads.
// Final h written to d_out every step; step t=63 leaves the answer.
// ---------------------------------------------------------------------------
__global__ __launch_bounds__(128) void lstm_auth_step_kernel(
    const bhalf* __restrict__ hpost, const bhalf* __restrict__ hin,
    const bhalf* __restrict__ kp, const bhalf* __restrict__ rp,
    const float* __restrict__ bias, float* __restrict__ cst,
    bhalf* __restrict__ hout, float* __restrict__ out, int t) {
  __shared__ float ls[4 * 256];
  int lane = threadIdx.x & 31;
  int q    = threadIdx.x >> 5;             // gate index 0..3
  int ncol = blockIdx.x;                   // 0..7 column groups of HA
  int nt   = q * (HAd / 16) + ncol;        // n-tile in 4HA space

  v8f c = {};
  const bhalf* abase = hpost + (long)t * HPd;  // row g -> hpost[(g*A + t)*HP]
#pragma unroll
  for (int ks = 0; ks < HPd / 32; ks += 2) {   // 8 k-steps
    v16bf a0 = load_a_bf16(abase, Ad * HPd, 0, ks * 32, lane);
    v16bf b0 = load_b_pack(kp, ks, nt, NG2 / 16, lane);
    v16bf a1 = load_a_bf16(abase, Ad * HPd, 0, ks * 32 + 32, lane);
    v16bf b1 = load_b_pack(kp, ks + 1, nt, NG2 / 16, lane);
    c = wmma_bf16(a0, b0, c);
    c = wmma_bf16(a1, b1, c);
  }
#pragma unroll
  for (int ks = 0; ks < HAd / 32; ks += 2) {   // 4 k-steps
    v16bf a0 = load_a_bf16(hin, HAd, 0, ks * 32, lane);
    v16bf b0 = load_b_pack(rp, ks, nt, NG2 / 16, lane);
    v16bf a1 = load_a_bf16(hin, HAd, 0, ks * 32 + 32, lane);
    v16bf b1 = load_b_pack(rp, ks + 1, nt, NG2 / 16, lane);
    c = wmma_bf16(a0, b0, c);
    c = wmma_bf16(a1, b1, c);
  }
  {
    int col   = lane & 15;
    int rbase = (lane >> 4) << 3;
#pragma unroll
    for (int j = 0; j < 8; ++j)
      ls[q * 256 + (rbase + j) * 16 + col] = c[j];
  }
  __syncthreads();
  for (int i = threadIdx.x; i < 256; i += 128) {
    int g_  = i >> 4;
    int col = i & 15;
    int n   = ncol * 16 + col;
    float gi = sigmoidf_(ls[0 * 256 + i] + bias[0 * HAd + n]);
    float gf = sigmoidf_(ls[1 * 256 + i] + bias[1 * HAd + n]);
    float gg = tanhf   (ls[2 * 256 + i] + bias[2 * HAd + n]);
    float go = sigmoidf_(ls[3 * 256 + i] + bias[3 * HAd + n]);
    long idx = (long)g_ * HAd + n;
    float cv = gf * cst[idx] + gi * gg;
    cst[idx]  = cv;
    float hv  = go * tanhf(cv);
    hout[idx] = (bhalf)hv;
    out[idx]  = hv;
  }
}

// ---------------------------------------------------------------------------
// Host launcher
// ---------------------------------------------------------------------------
extern "C" void kernel_launch(void* const* d_in, const int* in_sizes, int n_in,
                              void* d_out, int out_size, void* d_ws, size_t ws_size,
                              hipStream_t stream) {
  (void)in_sizes; (void)n_in; (void)out_size; (void)ws_size;
  const float* x       = (const float*)d_in[0];
  const float* w_embed = (const float*)d_in[1];
  const float* b_embed = (const float*)d_in[2];
  const float* k_post  = (const float*)d_in[3];
  const float* r_post  = (const float*)d_in[4];
  const float* b_post  = (const float*)d_in[5];
  const float* k_auth  = (const float*)d_in[6];
  const float* r_auth  = (const float*)d_in[7];
  const float* b_auth  = (const float*)d_in[8];
  float* out = (float*)d_out;

  // Workspace carve-out (256B-aligned slices)
  char* base = (char*)d_ws;
  size_t off = 0;
  auto carve = [&](size_t bytes) -> void* {
    off = (off + 255) & ~(size_t)255;
    void* p = base + off;
    off += bytes;
    return p;
  };
  bhalf* e_bf = (bhalf*)carve((size_t)ME * DEMB * sizeof(bhalf));
  bhalf* wep  = (bhalf*)carve((size_t)(DIN / 32) * (DEMB / 16) * 512 * sizeof(bhalf));
  bhalf* kpp  = (bhalf*)carve((size_t)(DEMB / 32) * (NG1 / 16) * 512 * sizeof(bhalf));
  bhalf* rpp  = (bhalf*)carve((size_t)(HPd / 32) * (NG1 / 16) * 512 * sizeof(bhalf));
  bhalf* kap  = (bhalf*)carve((size_t)(HPd / 32) * (NG2 / 16) * 512 * sizeof(bhalf));
  bhalf* rap  = (bhalf*)carve((size_t)(HAd / 32) * (NG2 / 16) * 512 * sizeof(bhalf));
  bhalf* hp0  = (bhalf*)carve((size_t)B1 * HPd * sizeof(bhalf));   // h ping
  bhalf* hp1  = (bhalf*)carve((size_t)B1 * HPd * sizeof(bhalf));   // h pong
  float* cpost= (float*)carve((size_t)B1 * HPd * sizeof(float));
  bhalf* ha0  = (bhalf*)carve((size_t)Gd * HAd * sizeof(bhalf));
  bhalf* ha1  = (bhalf*)carve((size_t)Gd * HAd * sizeof(bhalf));
  float* ca   = (float*)carve((size_t)Gd * HAd * sizeof(float));

  const int TB = 256;
  auto blocks = [](int threads) { return (threads + 255) / 256; };

  // Pack weights into bf16 WMMA B-fragment order
  pack_b_kernel<<<blocks((DIN / 32) * (DEMB / 16) * 32), TB, 0, stream>>>(w_embed, wep, DIN, DEMB);
  pack_b_kernel<<<blocks((DEMB / 32) * (NG1 / 16) * 32), TB, 0, stream>>>(k_post, kpp, DEMB, NG1);
  pack_b_kernel<<<blocks((HPd / 32) * (NG1 / 16) * 32), TB, 0, stream>>>(r_post, rpp, HPd, NG1);
  pack_b_kernel<<<blocks((HPd / 32) * (NG2 / 16) * 32), TB, 0, stream>>>(k_auth, kap, HPd, NG2);
  pack_b_kernel<<<blocks((HAd / 32) * (NG2 / 16) * 32), TB, 0, stream>>>(r_auth, rap, HAd, NG2);

  // Zero LSTM states (ping buffers only; pong is fully overwritten each step)
  init_state_kernel<<<blocks(B1 * HPd), TB, 0, stream>>>(hp0, cpost, ha0, ca);

  // Stage 1: embedding GEMM ([32768,768] x [768,384]) -> e_bf
  {
    int waves = (ME / 16) * ((DEMB / 16) / 4);   // 2048 * 6 = 12288
    embed_gemm_kernel<<<waves / 8, TB, 0, stream>>>(x, wep, b_embed, e_bf);
  }

  // Stage 2: post-level LSTM, 32 fused timestep kernels (h ping-pong)
  {
    dim3 grid((B1 / 16) * (HPd / 16));           // 64 * 16 = 1024 blocks
    for (int t = 0; t < Pd; ++t) {
      const bhalf* hin = (t & 1) ? hp1 : hp0;
      bhalf*      hout = (t & 1) ? hp0 : hp1;
      lstm_post_step_kernel<<<grid, 128, 0, stream>>>(e_bf, hin, kpp, rpp,
                                                      b_post, cpost, hout, t);
    }
  }
  const bhalf* hp_final = (Pd & 1) ? hp1 : hp0;  // Pd=32 -> hp0

  // Stage 3: author-level LSTM, 64 fused steps (final h -> d_out)
  {
    dim3 grid(HAd / 16);                         // 8 blocks
    for (int t = 0; t < Ad; ++t) {
      const bhalf* hin = (t & 1) ? ha1 : ha0;
      bhalf*      hout = (t & 1) ? ha0 : ha1;
      lstm_auth_step_kernel<<<grid, 128, 0, stream>>>(hp_final, hin, kap, rap,
                                                      b_auth, ca, hout, out, t);
    }
  }
}
